// DecisionTransformer_70282844832195
// MI455X (gfx1250) — matrix operand order repaired and checked by reading, not compiled, not run on hardware
//
#include <hip/hip_runtime.h>
#include <hip/hip_bf16.h>

// ---------------- problem constants ----------------
constexpr int kB  = 4;
constexpr int kL  = 512;
constexpr int kD  = 1024;
constexpr int kH  = 16;
constexpr int kFF = 4096;
constexpr int kNL = 4;
constexpr int kSS = 128;
constexpr int kA  = 16;
constexpr int kDH = kD / kH;     // 64
constexpr int kS  = 3 * kL;      // 1536
constexpr int kBS = kB * kS;     // 6144
constexpr int kS16 = kS / 16;    // 96
static_assert(kD == kH * kDH, "head split");

typedef __attribute__((ext_vector_type(16))) __bf16 v16bf;
typedef __attribute__((ext_vector_type(8)))  __bf16 v8bf;
typedef __attribute__((ext_vector_type(4)))  __bf16 v4bf;
typedef __attribute__((ext_vector_type(8)))  float  v8f;

#define WMMA_BF16(a, b, c) \
  __builtin_amdgcn_wmma_f32_16x16x32_bf16(false, (a), false, (b), (short)0, (c), false, false)

// ---- CDNA5 async global->LDS copy (CDNA5 ISA), tracked by ASYNCcnt ----
static __device__ inline void async_load_b128(unsigned lds_byte_off, const void* gaddr) {
  asm volatile("global_load_async_to_lds_b128 %0, %1, off"
               :: "v"(lds_byte_off),
                  "v"((unsigned long long)(size_t)gaddr)
               : "memory");
}
static __device__ inline void wait_asynccnt0() {
  asm volatile("s_wait_asynccnt 0x0" ::: "memory");
}

// ---- fragment loaders in contiguous k order (same for A and B; not the ISA table order) ----
// A (16x32 bf16): lane<16 -> row=lane, K {0..15}; lane>=16 -> row=lane-16, K {16..31}
// ld must be a multiple of 8 elements so each 8-elem chunk is 16B-aligned.
static __device__ inline v16bf load_a_frag(const __bf16* base, int ld) {
  int lane = threadIdx.x & 31;
  int row  = lane & 15;
  int kb   = (lane >> 4) << 4;          // 0 or 16
  const __bf16* p = base + (size_t)row * ld + kb;
  v8bf lo = *(const v8bf*)p;
  v8bf hi = *(const v8bf*)(p + 8);
  return __builtin_shufflevector(lo, hi, 0, 1, 2, 3, 4, 5, 6, 7,
                                 8, 9, 10, 11, 12, 13, 14, 15);
}
// B fragment from *swizzled* storage: wsw[((k/16)*N + col)*16 + (k%16)] so that
// each lane's 16 values (col fixed, K run of 16) are one contiguous 32B v16bf.
static __device__ inline v16bf load_b_swz(const __bf16* wsw, int kc0, int N, int col0) {
  int lane  = threadIdx.x & 31;
  int col   = lane & 15;
  int khalf = lane >> 4;                // 0 or 1 -> K rows kc0*16 .. kc0*16+31
  return *(const v16bf*)(wsw + ((size_t)(kc0 + khalf) * N + col0 + col) * 16);
}
// C (16x16 f32): VGPR j: lane<16 -> (m=j, n=lane); lane>=16 -> (m=j+8, n=lane-16)

// ---------------- weight swizzle passes (f32 row-major -> bf16 fragment layout) ----------------
__global__ __launch_bounds__(256) void swz_b_kernel(const float* __restrict__ W,
                                                    __bf16* __restrict__ out,
                                                    int K, int N) {
  size_t i = (size_t)blockIdx.x * 256 + threadIdx.x;
  if (i >= (size_t)K * N) return;
  int k = (int)(i / N);
  int c = (int)(i % N);
  out[((size_t)(k >> 4) * N + c) * 16 + (k & 15)] = (__bf16)W[i];
}
// per-head 64x64 swizzle for one (layer, q/k/v) weight block: W is H x 64 x 64
__global__ __launch_bounds__(256) void swz_qkv_kernel(const float* __restrict__ W,
                                                      __bf16* __restrict__ out) {
  int i = blockIdx.x * 256 + threadIdx.x;    // H*64*64 = 65536
  int h = i >> 12;
  int k = (i >> 6) & 63;
  int c = i & 63;
  out[((size_t)h << 12) + (((k >> 4) << 6) + c) * 16 + (k & 15)] = (__bf16)W[i];
}

// ---------------- embedding ----------------
__global__ __launch_bounds__(256) void embed_kernel(
    const float* __restrict__ reward, const float* __restrict__ state,
    const float* __restrict__ action, const float* __restrict__ timestep,
    const float* __restrict__ WR, const float* __restrict__ Ws,
    const float* __restrict__ Wa, const float* __restrict__ Wt,
    float* __restrict__ x, __bf16* __restrict__ xb) {
  int bl = blockIdx.x;                    // 0 .. B*L-1
  int b = bl / kL, l = bl % kL;
  __shared__ float st[kSS];
  for (int i = threadIdx.x; i < kSS; i += 256) st[i] = state[(size_t)bl * kSS + i];
  __syncthreads();
  float r = reward[bl], a = action[bl], t = timestep[bl];
  float sg = (t > 0.f) ? 1.f : ((t < 0.f) ? -1.f : 0.f);   // sign(t/D) == sign(t)
  size_t rowR = (size_t)b * kS + l;
  size_t rowS = rowR + kL;
  size_t rowA = rowR + 2 * kL;
  for (int d = threadIdx.x; d < kD; d += 256) {
    float pos = sg * Wt[d];
    float eR = r * WR[d] + pos;
    float acc = 0.f;
#pragma unroll 8
    for (int ss = 0; ss < kSS; ++ss) acc += st[ss] * Ws[(size_t)ss * kD + d];
    float eS = acc + pos;
    float eA = a * Wa[d] + pos;
    x[rowR * kD + d] = eR;  xb[rowR * kD + d] = (__bf16)eR;
    x[rowS * kD + d] = eS;  xb[rowS * kD + d] = (__bf16)eS;
    x[rowA * kD + d] = eA;  xb[rowA * kD + d] = (__bf16)eA;
  }
}

// ---------------- per-head QKV projection ----------------
// grid: (BS/128, H); block: 256 threads (8 waves), wave w handles rows r0+16w.
// Weights come pre-swizzled (fragment-native) from global (L2-resident); no LDS.
// q,k written row-major; v written in B-fragment swizzled layout for attention's PV.
__global__ __launch_bounds__(256) void qkv_kernel(
    const __bf16* __restrict__ xb, const __bf16* __restrict__ wsw,
    const float* __restrict__ bq, const float* __restrict__ bk,
    const float* __restrict__ bv,
    __bf16* __restrict__ q, __bf16* __restrict__ k, __bf16* __restrict__ vsw) {
  int h   = blockIdx.y;
  int tid = threadIdx.x;
  int wave = tid >> 5;
  int row  = blockIdx.x * 128 + wave * 16;
  const __bf16* abase = xb + (size_t)row * kD + h * kDH;
  v16bf a0 = load_a_frag(abase, kD);        // x cols 0..31 of head slice
  v16bf a1 = load_a_frag(abase + 32, kD);   // x cols 32..63
  int lane = tid & 31;
  int n    = lane & 15;
  int mb   = (lane >> 4) * 8;
  int b    = row / kS;
  int key16 = (row % kS) >> 4;              // 16-aligned tile => uniform over wave
  const float* bias[3] = {bq + h * kDH, bk + h * kDH, bv + h * kDH};
  __bf16* outp[2] = {q, k};
#pragma unroll
  for (int m = 0; m < 3; ++m) {
    const __bf16* wb = wsw + ((size_t)m * kH + h) * (kDH * kDH);
#pragma unroll
    for (int nt = 0; nt < 4; ++nt) {
      v16bf b0 = load_b_swz(wb, 0, kDH, nt * 16);  // K rows 0..31
      v16bf b1 = load_b_swz(wb, 2, kDH, nt * 16);  // K rows 32..63
      v8f c = {};
      c = WMMA_BF16(a0, b0, c);
      c = WMMA_BF16(a1, b1, c);
      float bi = bias[m][nt * 16 + n];
      if (m < 2) {
#pragma unroll
        for (int j = 0; j < 8; ++j)
          outp[m][(size_t)(row + mb + j) * kD + h * kDH + nt * 16 + n] =
              (__bf16)(c[j] + bi);
      } else {
        // v: swizzled store, 8 consecutive keys -> one contiguous 16B run
        v8bf pk;
#pragma unroll
        for (int j = 0; j < 8; ++j) pk[j] = (__bf16)(c[j] + bi);
        size_t run = ((size_t)(b * kH + h) * kS16 + key16) * kDH + nt * 16 + n;
        *(v8bf*)(vsw + run * 16 + mb) = pk;
      }
    }
  }
}

// ---------------- flash attention, one wave per (b,h,16-query tile) ----------------
__global__ __launch_bounds__(32) void attn_kernel(
    const __bf16* __restrict__ q, const __bf16* __restrict__ kk,
    const __bf16* __restrict__ vsw, __bf16* __restrict__ ob) {
  __shared__ __bf16 Ps[16][32];
  int qt = blockIdx.x;
  int bh = blockIdx.y;
  int b = bh >> 4, h = bh & 15;
  int q0 = qt * 16;
  size_t base = (size_t)b * kS * kD + (size_t)h * kDH;
  const __bf16* vbase = vsw + (size_t)bh * kS16 * kDH * 16;
  int lane  = threadIdx.x & 31;
  int n     = lane & 15;
  int mb    = (lane >> 4) * 8;
  int kbh   = (lane >> 4) << 4;          // K-half select for QK^T B frags
  int khalf = lane >> 4;

  const __bf16* qbase = q + base + (size_t)q0 * kD;
  v16bf qa0 = load_a_frag(qbase, kD);
  v16bf qa1 = load_a_frag(qbase + 32, kD);

  v8f oacc[4] = {};
  float mi[8], li[8];
#pragma unroll
  for (int j = 0; j < 8; ++j) { mi[j] = -1e30f; li[j] = 0.f; }

  for (int kt = 0; kt < kS; kt += 32) {
    // ---- S = Q K^T for 32 keys; K rows are contiguous bf16 runs -> v16bf loads ----
    const __bf16* kp0 = kk + base + (size_t)(kt + n) * kD + kbh;
    const __bf16* kp1 = kk + base + (size_t)(kt + 16 + n) * kD + kbh;
    v8f s0 = {}, s1 = {};
    {
      v16bf bA = *(const v16bf*)kp0;
      v16bf bB = *(const v16bf*)(kp0 + 32);
      s0 = WMMA_BF16(qa0, bA, s0);
      s0 = WMMA_BF16(qa1, bB, s0);
      v16bf bC = *(const v16bf*)kp1;
      v16bf bD = *(const v16bf*)(kp1 + 32);
      s1 = WMMA_BF16(qa0, bC, s1);
      s1 = WMMA_BF16(qa1, bD, s1);
    }
    // ---- mask: only the last query row masks keys >= S-L ----
#pragma unroll
    for (int j = 0; j < 8; ++j) {
      int qrow = q0 + mb + j;
      if (qrow == kS - 1) {
        if (kt + n      >= kS - kL) s0[j] = -1e30f;
        if (kt + 16 + n >= kS - kL) s1[j] = -1e30f;
      }
    }
    // ---- online softmax ----
    float mnew[8], scl[8];
#pragma unroll
    for (int j = 0; j < 8; ++j) {
      float t = fmaxf(s0[j], s1[j]);
      t = fmaxf(t, __shfl_xor(t, 1, 32));
      t = fmaxf(t, __shfl_xor(t, 2, 32));
      t = fmaxf(t, __shfl_xor(t, 4, 32));
      t = fmaxf(t, __shfl_xor(t, 8, 32));
      mnew[j] = fmaxf(mi[j], t);
      scl[j]  = __expf(mi[j] - mnew[j]);
      mi[j]   = mnew[j];
    }
    v8f p0, p1;
#pragma unroll
    for (int j = 0; j < 8; ++j) {
      p0[j] = __expf(s0[j] - mnew[j]);
      p1[j] = __expf(s1[j] - mnew[j]);
      float t = p0[j] + p1[j];
      t += __shfl_xor(t, 1, 32);
      t += __shfl_xor(t, 2, 32);
      t += __shfl_xor(t, 4, 32);
      t += __shfl_xor(t, 8, 32);
      li[j] = li[j] * scl[j] + t;
      oacc[0][j] *= scl[j]; oacc[1][j] *= scl[j];
      oacc[2][j] *= scl[j]; oacc[3][j] *= scl[j];
    }
    // ---- P: C layout -> A layout via LDS ----
    __syncthreads();
#pragma unroll
    for (int j = 0; j < 8; ++j) {
      Ps[mb + j][n]      = (__bf16)p0[j];
      Ps[mb + j][16 + n] = (__bf16)p1[j];
    }
    __syncthreads();
    v16bf pa = load_a_frag(&Ps[0][0], 32);
    // ---- O += P V : V fragments are single contiguous v16bf from swizzled V ----
    int kt16 = (kt >> 4) + khalf;
#pragma unroll
    for (int nt = 0; nt < 4; ++nt) {
      v16bf vb = *(const v16bf*)(vbase + ((size_t)kt16 * kDH + nt * 16 + n) * 16);
      oacc[nt] = WMMA_BF16(pa, vb, oacc[nt]);
    }
  }
  // ---- normalize (reference scales AFTER softmax by 1/sqrt(D) = 1/32) ----
  const float isc = 0.03125f;
#pragma unroll
  for (int nt = 0; nt < 4; ++nt)
#pragma unroll
    for (int j = 0; j < 8; ++j) {
      float val = oacc[nt][j] / li[j] * isc;
      ob[base + (size_t)(q0 + mb + j) * kD + nt * 16 + n] = (__bf16)val;
    }
}

// ---------------- tiled WMMA GEMM: C = A(bf16 MxK) @ Bsw(swizzled bf16 KxN) + bias ----------------
// block 256 thr = 8 waves as 4(m) x 2(n); block tile 128x128, wave tile 32x64, BK=32.
// A tiles stream global->LDS via CDNA5 async-to-LDS (ASYNCcnt), double buffered,
// one barrier per k-iteration. B fragments load directly from swizzled global (L2).
template <int RELU, int OUTB>
__global__ __launch_bounds__(256) void gemm_bf16_kernel(
    const __bf16* __restrict__ Am, const __bf16* __restrict__ Bsw,
    const float* __restrict__ bias, float* __restrict__ Cf,
    __bf16* __restrict__ Cb, int M, int N, int K) {
  constexpr int BM = 128, BK = 32, LDA = BK + 8;
  __shared__ __bf16 As[2][BM][LDA];
  int bm = blockIdx.x * BM;
  int bn = blockIdx.y * 128;
  int tid  = threadIdx.x;
  int wave = tid >> 5;
  int wm = (wave & 3) * 32;
  int wn = (wave >> 2) * 64;
  int sr = tid >> 1;                 // staging row 0..127
  int sc = (tid & 1) * 16;           // staging col 0 or 16
  const char* lds0 = (const char*)&As[0][0][0];
  unsigned d0 = (unsigned)((const char*)&As[0][sr][sc] - lds0);
  unsigned d1 = (unsigned)((const char*)&As[1][sr][sc] - lds0);
  const __bf16* srow = Am + (size_t)(bm + sr) * K + sc;
  v8f c[2][4] = {};
  // prologue: async-stage tile 0 into buffer 0
  async_load_b128(d0,      srow);
  async_load_b128(d0 + 16, srow + 8);
  int bi = 0;
  for (int k0 = 0; k0 < K; k0 += BK) {
    wait_asynccnt0();          // own async stores to LDS complete
    __syncthreads();           // everyone's complete; also protects buffer reuse
    if (k0 + BK < K) {         // async-stage next tile into the other buffer
      unsigned dn = bi ? d0 : d1;
      async_load_b128(dn,      srow + (k0 + BK));
      async_load_b128(dn + 16, srow + (k0 + BK) + 8);
      __builtin_prefetch(Bsw + ((size_t)((k0 + BK) >> 4) * N + bn + (tid >> 1)) * 16, 0, 1);
    }
    v16bf a0 = load_a_frag(&As[bi][wm][0],      LDA);
    v16bf a1 = load_a_frag(&As[bi][wm + 16][0], LDA);
    int kc = k0 >> 4;
#pragma unroll
    for (int nt = 0; nt < 4; ++nt) {
      v16bf bfrag = load_b_swz(Bsw, kc, N, bn + wn + nt * 16);
      c[0][nt] = WMMA_BF16(a0, bfrag, c[0][nt]);
      c[1][nt] = WMMA_BF16(a1, bfrag, c[1][nt]);
    }
    bi ^= 1;
  }
  int lane = tid & 31;
  int n  = lane & 15;
  int mb = (lane >> 4) * 8;
#pragma unroll
  for (int nt = 0; nt < 4; ++nt) {
    int col = bn + wn + nt * 16 + n;
    float bv = bias[col];
#pragma unroll
    for (int mt = 0; mt < 2; ++mt) {
#pragma unroll
      for (int j = 0; j < 8; ++j) {
        int row = bm + wm + mt * 16 + mb + j;
        float vvv = c[mt][nt][j] + bv;
        if (RELU) vvv = vvv > 0.f ? vvv : 0.f;
        if (OUTB) Cb[(size_t)row * N + col] = (__bf16)vvv;
        else      Cf[(size_t)row * N + col] = vvv;
      }
    }
  }
}

// ---------------- fused residual + LayerNorm; refreshes bf16 mirror ----------------
__global__ __launch_bounds__(256) void ln_kernel(
    float* __restrict__ x, const float* __restrict__ y,
    const float* __restrict__ g, const float* __restrict__ bta,
    __bf16* __restrict__ xb) {
  int row = blockIdx.x;
  int tid = threadIdx.x;
  __shared__ float red[2][8];
  __shared__ float mu_sh, rstd_sh;
  float4 vx = *(const float4*)(x + (size_t)row * kD + tid * 4);
  float4 vy = *(const float4*)(y + (size_t)row * kD + tid * 4);
  float lv[4] = {vx.x + vy.x, vx.y + vy.y, vx.z + vy.z, vx.w + vy.w};
  float s = lv[0] + lv[1] + lv[2] + lv[3];
  float s2 = lv[0] * lv[0] + lv[1] * lv[1] + lv[2] * lv[2] + lv[3] * lv[3];
  for (int m = 1; m < 32; m <<= 1) {
    s  += __shfl_xor(s,  m, 32);
    s2 += __shfl_xor(s2, m, 32);
  }
  if ((tid & 31) == 0) { red[0][tid >> 5] = s; red[1][tid >> 5] = s2; }
  __syncthreads();
  if (tid == 0) {
    float S1 = 0.f, S2 = 0.f;
    for (int w = 0; w < 8; ++w) { S1 += red[0][w]; S2 += red[1][w]; }
    float mu = S1 / kD;
    mu_sh = mu;
    rstd_sh = rsqrtf(S2 / kD - mu * mu + 1e-5f);
  }
  __syncthreads();
  float mu = mu_sh, rstd = rstd_sh;
  float4 go = *(const float4*)(g + tid * 4);
  float4 bo = *(const float4*)(bta + tid * 4);
  float4 outv;
  outv.x = (lv[0] - mu) * rstd * go.x + bo.x;
  outv.y = (lv[1] - mu) * rstd * go.y + bo.y;
  outv.z = (lv[2] - mu) * rstd * go.z + bo.z;
  outv.w = (lv[3] - mu) * rstd * go.w + bo.w;
  *(float4*)(x + (size_t)row * kD + tid * 4) = outv;
  v4bf pk = {(__bf16)outv.x, (__bf16)outv.y, (__bf16)outv.z, (__bf16)outv.w};
  *(v4bf*)(xb + (size_t)row * kD + tid * 4) = pk;
}

// ---------------- tiny action head ----------------
__global__ __launch_bounds__(256) void head_kernel(
    const float* __restrict__ x, const float* __restrict__ Wa,
    const float* __restrict__ ba, float* __restrict__ out) {
  int idx = blockIdx.x * 256 + threadIdx.x;     // B*L*A = 32768
  if (idx >= kB * kL * kA) return;
  int a = idx & (kA - 1);
  int bl = idx / kA;
  int b = bl / kL, l = bl % kL;
  const float* xr = x + ((size_t)b * kS + 2 * kL + l) * kD;
  float acc = ba[a];
#pragma unroll 8
  for (int d = 0; d < kD; ++d) acc += xr[d] * Wa[(size_t)d * kA + a];
  out[idx] = acc;
}

// ---------------- host launcher ----------------
extern "C" void kernel_launch(void* const* d_in, const int* in_sizes, int n_in,
                              void* d_out, int out_size, void* d_ws, size_t ws_size,
                              hipStream_t stream) {
  (void)in_sizes; (void)n_in; (void)out_size; (void)ws_size;
  const float* reward   = (const float*)d_in[0];
  const float* state    = (const float*)d_in[1];
  const float* action   = (const float*)d_in[2];
  const float* timestep = (const float*)d_in[3];
  const float* eRW = (const float*)d_in[4];
  const float* esW = (const float*)d_in[5];
  const float* eaW = (const float*)d_in[6];
  const float* etW = (const float*)d_in[7];
  const float* Wq = (const float*)d_in[8];
  const float* bq = (const float*)d_in[9];
  const float* Wk = (const float*)d_in[10];
  const float* bk = (const float*)d_in[11];
  const float* Wv = (const float*)d_in[12];
  const float* bv = (const float*)d_in[13];
  const float* Wo = (const float*)d_in[14];
  const float* bo = (const float*)d_in[15];
  const float* ln1g = (const float*)d_in[16];
  const float* ln1b = (const float*)d_in[17];
  const float* W1 = (const float*)d_in[18];
  const float* b1 = (const float*)d_in[19];
  const float* W2 = (const float*)d_in[20];
  const float* b2 = (const float*)d_in[21];
  const float* ln2g = (const float*)d_in[22];
  const float* ln2b = (const float*)d_in[23];
  const float* actW = (const float*)d_in[24];
  const float* actb = (const float*)d_in[25];

  char* ws = (char*)d_ws;
  size_t off = 0;
  auto alloc = [&](size_t bytes) {
    void* p = ws + off;
    off += (bytes + 255) & ~(size_t)255;
    return p;
  };
  float*  x   = (float*)alloc((size_t)kBS * kD * 4);
  __bf16* xb  = (__bf16*)alloc((size_t)kBS * kD * 2);
  __bf16* qb  = (__bf16*)alloc((size_t)kBS * kD * 2);
  __bf16* kb  = (__bf16*)alloc((size_t)kBS * kD * 2);
  __bf16* vsw = (__bf16*)alloc((size_t)kBS * kD * 2);
  __bf16* ob  = (__bf16*)alloc((size_t)kBS * kD * 2);
  float*  y   = (float*)alloc((size_t)kBS * kD * 4);
  __bf16* hb  = (__bf16*)alloc((size_t)kBS * kFF * 2);
  __bf16* wqkvsw = (__bf16*)alloc((size_t)kNL * 3 * kH * kDH * kDH * 2);
  __bf16* wosw   = (__bf16*)alloc((size_t)kNL * kD * kD * 2);
  __bf16* w1sw   = (__bf16*)alloc((size_t)kNL * kD * kFF * 2);
  __bf16* w2sw   = (__bf16*)alloc((size_t)kNL * kFF * kD * 2);

  // ---- one-time weight swizzle (fragment-native bf16 layouts) ----
  const int hdblk = kH * kDH * kDH;          // 65536
  for (int l = 0; l < kNL; ++l) {
    swz_qkv_kernel<<<hdblk / 256, 256, 0, stream>>>(Wq + (size_t)l * hdblk,
                                                    wqkvsw + ((size_t)l * 3 + 0) * hdblk);
    swz_qkv_kernel<<<hdblk / 256, 256, 0, stream>>>(Wk + (size_t)l * hdblk,
                                                    wqkvsw + ((size_t)l * 3 + 1) * hdblk);
    swz_qkv_kernel<<<hdblk / 256, 256, 0, stream>>>(Wv + (size_t)l * hdblk,
                                                    wqkvsw + ((size_t)l * 3 + 2) * hdblk);
    swz_b_kernel<<<(kD * kD) / 256, 256, 0, stream>>>(Wo + (size_t)l * kD * kD,
                                                      wosw + (size_t)l * kD * kD, kD, kD);
    swz_b_kernel<<<(kD * kFF) / 256, 256, 0, stream>>>(W1 + (size_t)l * kD * kFF,
                                                       w1sw + (size_t)l * kD * kFF, kD, kFF);
    swz_b_kernel<<<(kFF * kD) / 256, 256, 0, stream>>>(W2 + (size_t)l * kFF * kD,
                                                       w2sw + (size_t)l * kFF * kD, kFF, kD);
  }

  embed_kernel<<<kB * kL, 256, 0, stream>>>(reward, state, action, timestep,
                                            eRW, esW, eaW, etW, x, xb);
  for (int l = 0; l < kNL; ++l) {
    qkv_kernel<<<dim3(kBS / 128, kH), 256, 0, stream>>>(
        xb, wqkvsw + (size_t)l * 3 * hdblk,
        bq + (size_t)l * kH * kDH, bk + (size_t)l * kH * kDH, bv + (size_t)l * kH * kDH,
        qb, kb, vsw);
    attn_kernel<<<dim3(kS / 16, kB * kH), 32, 0, stream>>>(qb, kb, vsw, ob);
    gemm_bf16_kernel<0, 0><<<dim3(kBS / 128, kD / 128), 256, 0, stream>>>(
        ob, wosw + (size_t)l * kD * kD, bo + (size_t)l * kD, y, nullptr, kBS, kD, kD);
    ln_kernel<<<kBS, 256, 0, stream>>>(x, y, ln1g + (size_t)l * kD,
                                       ln1b + (size_t)l * kD, xb);
    gemm_bf16_kernel<1, 1><<<dim3(kBS / 128, kFF / 128), 256, 0, stream>>>(
        xb, w1sw + (size_t)l * kD * kFF, b1 + (size_t)l * kFF, nullptr, hb, kBS, kFF, kD);
    gemm_bf16_kernel<0, 0><<<dim3(kBS / 128, kD / 128), 256, 0, stream>>>(
        hb, w2sw + (size_t)l * kFF * kD, b2 + (size_t)l * kD, y, nullptr, kBS, kD, kFF);
    ln_kernel<<<kBS, 256, 0, stream>>>(x, y, ln2g + (size_t)l * kD,
                                       ln2b + (size_t)l * kD, xb);
  }
  head_kernel<<<(kB * kL * kA + 255) / 256, 256, 0, stream>>>(x, actW, actb,
                                                              (float*)d_out);
}
